// DTRN_20727512170833
// MI455X (gfx1250) — compile-verified, matrix operand order checked by reading, not run on hardware
//
#include <hip/hip_runtime.h>
#include <hip/hip_bf16.h>

typedef __bf16 bf16;
typedef __attribute__((ext_vector_type(16))) __bf16 v16bf;
typedef __attribute__((ext_vector_type(8)))  __bf16 v8bf;
typedef __attribute__((ext_vector_type(8)))  float  v8f;

#define B_ 16
#define T_ 2048
#define H_ 256
#define G_ 1024   // 4*H

// ---------------------------------------------------------------------------
// fast transcendentals (v_exp_f32 / v_rcp_f32)
// ---------------------------------------------------------------------------
__device__ __forceinline__ float sigf(float x) {
    // sigmoid(x) = 1 / (1 + 2^(-x*log2(e)))
    return __builtin_amdgcn_rcpf(1.0f + __builtin_amdgcn_exp2f(-1.44269504f * x));
}
__device__ __forceinline__ float tanhfast(float x) {
    return __builtin_fmaf(2.0f, sigf(2.0f * x), -1.0f);
}

// ---------------------------------------------------------------------------
// Whh [4H, H] f32  ->  bf16 fragment-major buffer [tile(64)][ks(8)][lane(32)][16]
// B-fragment element e of lane L in k-step ks of tile tn holds
//   Whh[n, k],  n = tn*16 + (L&15),  k = ks*32 + (L>>4)*8 + (e<8 ? e : 16+e-8)
// ---------------------------------------------------------------------------
__global__ void DTRN_swizzle(const float* __restrict__ Whh, bf16* __restrict__ frag) {
    int id = blockIdx.x * blockDim.x + threadIdx.x;     // 64*8*32 = 16384
    if (id >= 64 * 8 * 32) return;
    int lane = id & 31;
    int ks   = (id >> 5) & 7;
    int tile = id >> 8;
    int n    = tile * 16 + (lane & 15);
    int kb   = (lane >> 4) * 8;
    bf16* dst = frag + (size_t)id * 16;
    const float* src = Whh + (size_t)n * H_ + ks * 32 + kb;
#pragma unroll
    for (int e = 0; e < 8; ++e) dst[e]     = (bf16)src[e];
#pragma unroll
    for (int e = 0; e < 8; ++e) dst[8 + e] = (bf16)src[16 + e];
}

// ---------------------------------------------------------------------------
// xp = X @ Wih^T + (bih + bhh), stored bf16 [M, 1024].
// X: [M, K] f32 (M = B*T = 32768), Wih: [1024, K] f32.
// One wave per 16x16 output tile, WMMA bf16, K-step 32.
// ---------------------------------------------------------------------------
__global__ __launch_bounds__(256)
void DTRN_gemm_xp(const float* __restrict__ X, const float* __restrict__ Wih,
                  const float* __restrict__ bih, const float* __restrict__ bhh,
                  bf16* __restrict__ xp, int K) {
    const int wave  = threadIdx.x >> 5;
    const int lane  = threadIdx.x & 31;
    const int jloc  = lane & 15;
    const int khalf = lane >> 4;
    const int tm = blockIdx.x;                 // 0..2047
    const int tn = blockIdx.y * 8 + wave;      // 0..63
    const int rowA = tm * 16 + jloc;
    const int nB   = tn * 16 + jloc;
    const float* ar = X   + (size_t)rowA * K + khalf * 8;
    const float* br = Wih + (size_t)nB   * K + khalf * 8;

    v8f acc = {};
    for (int k0 = 0; k0 < K; k0 += 32) {
        float af[16], bfl[16];
        *(float4*)(af  + 0)  = *(const float4*)(ar + k0);
        *(float4*)(af  + 4)  = *(const float4*)(ar + k0 + 4);
        *(float4*)(af  + 8)  = *(const float4*)(ar + k0 + 16);
        *(float4*)(af  + 12) = *(const float4*)(ar + k0 + 20);
        *(float4*)(bfl + 0)  = *(const float4*)(br + k0);
        *(float4*)(bfl + 4)  = *(const float4*)(br + k0 + 4);
        *(float4*)(bfl + 8)  = *(const float4*)(br + k0 + 16);
        *(float4*)(bfl + 12) = *(const float4*)(br + k0 + 20);
        v16bf a, b;
#pragma unroll
        for (int e = 0; e < 16; ++e) { a[e] = (bf16)af[e]; b[e] = (bf16)bfl[e]; }
        acc = __builtin_amdgcn_wmma_f32_16x16x32_bf16(false, a, false, b,
                                                      (short)0, acc, false, false);
    }
    const float bias = bih[nB] + bhh[nB];
    const int mb = tm * 16 + khalf * 8;
#pragma unroll
    for (int r = 0; r < 8; ++r)
        xp[(size_t)(mb + r) * G_ + nB] = (bf16)(acc[r] + bias);
}

// ---------------------------------------------------------------------------
// coefficient kernel: cf[b,s], cb[b,s] in scan order, s==0 zeroed.
//   cf[b,s] = 0.9*sigmoid(dot(dof[b,s,:],   cw_f) + cb_f)
//   cb[b,s] = 0.9*sigmoid(dot(dob[b,T-1-s,:],cw_b) + cb_b)
// ---------------------------------------------------------------------------
__global__ void DTRN_coeff(const float* __restrict__ dof, const float* __restrict__ dob,
                           const float* __restrict__ cw_f, const float* __restrict__ cb_f,
                           const float* __restrict__ cw_b, const float* __restrict__ cb_b,
                           float* __restrict__ cfA, float* __restrict__ cbA) {
    int id = blockIdx.x * blockDim.x + threadIdx.x;     // B*T
    if (id >= B_ * T_) return;
    int b = id / T_, s = id % T_;
    const float4* wf4 = (const float4*)cw_f;
    const float4* wb4 = (const float4*)cw_b;
    const float4* hf4 = (const float4*)(dof + ((size_t)b * T_ + s) * H_);
    const float4* hb4 = (const float4*)(dob + ((size_t)b * T_ + (T_ - 1 - s)) * H_);
    float af = 0.f, ab = 0.f;
    for (int q = 0; q < H_ / 4; ++q) {
        float4 h = hf4[q], w = wf4[q];
        af += h.x * w.x + h.y * w.y + h.z * w.z + h.w * w.w;
        float4 h2 = hb4[q], w2 = wb4[q];
        ab += h2.x * w2.x + h2.y * w2.y + h2.z * w2.z + h2.w * w2.w;
    }
    cfA[id] = (s == 0) ? 0.f : 0.9f * sigf(af + cb_f[0]);
    cbA[id] = (s == 0) ? 0.f : 0.9f * sigf(ab + cb_b[0]);
}

// ---------------------------------------------------------------------------
// Persistent LSTM scan. grid.x = 2 (dir 0 = forward, 1 = backward), 512 thr.
// Per step: g = h @ Whh^T via 16 waves x 32 v_wmma_f32_16x16x32_bf16
// (wave w owns hidden units j in [16w,16w+16) across all four gates, so the
//  gate combine + nonlinearity stays in registers; c/h state lives in regs,
//  bf16 h for the A-operand lives in 8 KB LDS).
// ---------------------------------------------------------------------------
__global__ __launch_bounds__(512, 1)
void DTRN_scan(const bf16* __restrict__ xp_f, const bf16* __restrict__ xp_b,
               const bf16* __restrict__ wfrag_f, const bf16* __restrict__ wfrag_b,
               const float* __restrict__ coef_f, const float* __restrict__ coef_b,
               float* __restrict__ out_f, float* __restrict__ out_b, int out_stride) {
    const int dir = blockIdx.x;
    const bf16*  xp    = dir ? xp_b    : xp_f;
    const bf16*  wfrag = dir ? wfrag_b : wfrag_f;
    const float* coef  = dir ? coef_b  : coef_f;
    float*       out   = dir ? out_b   : out_f;

    const int tid   = threadIdx.x;
    const int wave  = tid >> 5;        // 0..15
    const int lane  = tid & 31;
    const int jloc  = lane & 15;
    const int khalf = lane >> 4;       // A K-half / D M-half selector
    const int j     = wave * 16 + jloc;   // hidden unit this thread owns
    const int mbase = khalf * 8;          // D rows: mbase..mbase+7

    __shared__ __align__(16) bf16 h_bf[B_][H_];   // 8 KB previous hidden (A operand)
    for (int i = tid; i < B_ * H_; i += blockDim.x) (&h_bf[0][0])[i] = (bf16)0.0f;
    float c_reg[8], h_reg[8];
#pragma unroll
    for (int r = 0; r < 8; ++r) { c_reg[r] = 0.f; h_reg[r] = 0.f; }
    __syncthreads();

    const v16bf* wf   = (const v16bf*)wfrag;
    const v8bf*  hrow = (const v8bf*)(&h_bf[jloc][0]);  // A row M = lane&15

    for (int s = 0; s < T_; ++s) {
        const int t = dir ? (T_ - 1 - s) : s;

        v8f acc_i = {}, acc_ff = {}, acc_g = {}, acc_o = {};
#pragma unroll
        for (int ks = 0; ks < 8; ++ks) {
            union { v16bf v; v8bf h2[2]; } a;
            const int g8 = ks * 4 + khalf;           // 8-elem group index in h row
            a.h2[0] = hrow[g8];
            a.h2[1] = hrow[g8 + 2];
            const size_t fb = ((size_t)wave * 8 + ks) * 32 + lane;
            v16bf bi = wf[fb];                       // gate i : tile = wave
            v16bf bf2 = wf[fb + 1 * 4096];           // gate f : tile = wave+16
            v16bf bg = wf[fb + 2 * 4096];            // gate g : tile = wave+32
            v16bf bo = wf[fb + 3 * 4096];            // gate o : tile = wave+48
            acc_i  = __builtin_amdgcn_wmma_f32_16x16x32_bf16(false, a.v, false, bi,  (short)0, acc_i,  false, false);
            acc_ff = __builtin_amdgcn_wmma_f32_16x16x32_bf16(false, a.v, false, bf2, (short)0, acc_ff, false, false);
            acc_g  = __builtin_amdgcn_wmma_f32_16x16x32_bf16(false, a.v, false, bg,  (short)0, acc_g,  false, false);
            acc_o  = __builtin_amdgcn_wmma_f32_16x16x32_bf16(false, a.v, false, bo,  (short)0, acc_o,  false, false);
        }
        __syncthreads();   // all waves done reading h_bf

#pragma unroll
        for (int r = 0; r < 8; ++r) {
            const int b = mbase + r;
            const bf16* xr = xp + ((size_t)b * T_ + t) * (size_t)G_;
            float ii = acc_i[r]  + (float)xr[j];
            float ff = acc_ff[r] + (float)xr[j + 256];
            float gg = acc_g[r]  + (float)xr[j + 512];
            float oo = acc_o[r]  + (float)xr[j + 768];
            float cf = coef ? coef[(size_t)b * T_ + s] : 0.0f;
            float cn = sigf(ff) * c_reg[r] + sigf(ii) * tanhfast(gg);
            float hn = sigf(oo) * tanhfast(cn) + cf * h_reg[r];
            c_reg[r] = cn;
            h_reg[r] = hn;
            h_bf[b][j] = (bf16)hn;
            out[((size_t)b * T_ + t) * (size_t)out_stride + j] = hn;
        }
        __syncthreads();   // h_bf fully updated before next step's GEMM
    }
}

// ---------------------------------------------------------------------------
// host-side launch
// ---------------------------------------------------------------------------
extern "C" void kernel_launch(void* const* d_in, const int* in_sizes, int n_in,
                              void* d_out, int out_size, void* d_ws, size_t ws_size,
                              hipStream_t stream) {
    (void)in_sizes; (void)n_in; (void)out_size; (void)ws_size;
    const float* inputs  = (const float*)d_in[0];
    const float* dWih_f  = (const float*)d_in[1];
    const float* dWhh_f  = (const float*)d_in[2];
    const float* dbih_f  = (const float*)d_in[3];
    const float* dbhh_f  = (const float*)d_in[4];
    const float* dWih_b  = (const float*)d_in[5];
    const float* dWhh_b  = (const float*)d_in[6];
    const float* dbih_b  = (const float*)d_in[7];
    const float* dbhh_b  = (const float*)d_in[8];
    const float* l0Wih_f = (const float*)d_in[9];
    const float* l0Whh_f = (const float*)d_in[10];
    const float* l0bih_f = (const float*)d_in[11];
    const float* l0bhh_f = (const float*)d_in[12];
    const float* l0Wih_b = (const float*)d_in[13];
    const float* l0Whh_b = (const float*)d_in[14];
    const float* l0bih_b = (const float*)d_in[15];
    const float* l0bhh_b = (const float*)d_in[16];
    const float* l1Wih_f = (const float*)d_in[17];
    const float* l1Whh_f = (const float*)d_in[18];
    const float* l1bih_f = (const float*)d_in[19];
    const float* l1bhh_f = (const float*)d_in[20];
    const float* l1Wih_b = (const float*)d_in[21];
    const float* l1Whh_b = (const float*)d_in[22];
    const float* l1bih_b = (const float*)d_in[23];
    const float* l1bhh_b = (const float*)d_in[24];
    const float* cw_f    = (const float*)d_in[25];
    const float* cb_f    = (const float*)d_in[26];
    const float* cw_b    = (const float*)d_in[27];
    const float* cb_b    = (const float*)d_in[28];

    char* ws = (char*)d_ws;
    // workspace layout (bytes)
    bf16*  xp_f = (bf16*)(ws);                         //  64 MB  [32768,1024] bf16
    bf16*  xp_b = (bf16*)(ws + (64ull  << 20));        //  64 MB
    float* dof  = (float*)(ws + (128ull << 20));       //  32 MB  [B,T,H] f32
    float* dob  = (float*)(ws + (160ull << 20));       //  32 MB
    float* x1   = (float*)(ws + (128ull << 20));       //  64 MB  overlays dof/dob (consumed first)
    float* cfA  = (float*)(ws + (192ull << 20));       // 128 KB
    float* cbA  = (float*)(ws + (193ull << 20));       // 128 KB
    bf16*  wfrag = (bf16*)(ws + (194ull << 20));       // 6 x 512 KB
    const size_t WFN = (size_t)G_ * H_;                // 262144 elems per matrix
    bf16 *wf_df = wfrag,           *wf_db = wfrag + WFN,
         *wf_0f = wfrag + 2 * WFN, *wf_0b = wfrag + 3 * WFN,
         *wf_1f = wfrag + 4 * WFN, *wf_1b = wfrag + 5 * WFN;

    // 1) swizzle all recurrent weights into WMMA fragment order (bf16)
    dim3 swb(256), swg(64);
    DTRN_swizzle<<<swg, swb, 0, stream>>>(dWhh_f,  wf_df);
    DTRN_swizzle<<<swg, swb, 0, stream>>>(dWhh_b,  wf_db);
    DTRN_swizzle<<<swg, swb, 0, stream>>>(l0Whh_f, wf_0f);
    DTRN_swizzle<<<swg, swb, 0, stream>>>(l0Whh_b, wf_0b);
    DTRN_swizzle<<<swg, swb, 0, stream>>>(l1Whh_f, wf_1f);
    DTRN_swizzle<<<swg, swb, 0, stream>>>(l1Whh_b, wf_1b);

    dim3 gb(256), gg(2048, 8);   // 32768/16 tile-rows x (64 tile-cols / 8 waves)

    // 2) discount-generator layer
    DTRN_gemm_xp<<<gg, gb, 0, stream>>>(inputs, dWih_f, dbih_f, dbhh_f, xp_f, 256);
    DTRN_gemm_xp<<<gg, gb, 0, stream>>>(inputs, dWih_b, dbih_b, dbhh_b, xp_b, 256);
    DTRN_scan<<<2, 512, 0, stream>>>(xp_f, xp_b, wf_df, wf_db,
                                     nullptr, nullptr, dof, dob, H_);

    // 3) coefficients
    DTRN_coeff<<<(B_ * T_ + 255) / 256, 256, 0, stream>>>(dof, dob, cw_f, cb_f,
                                                          cw_b, cb_b, cfA, cbA);

    // 4) layer 0 (input -> x1 = concat(of, ob))
    DTRN_gemm_xp<<<gg, gb, 0, stream>>>(inputs, l0Wih_f, l0bih_f, l0bhh_f, xp_f, 256);
    DTRN_gemm_xp<<<gg, gb, 0, stream>>>(inputs, l0Wih_b, l0bih_b, l0bhh_b, xp_b, 256);
    DTRN_scan<<<2, 512, 0, stream>>>(xp_f, xp_b, wf_0f, wf_0b,
                                     cfA, cbA, x1, x1 + H_, 2 * H_);

    // 5) layer 1 (x1 -> d_out), K = 512
    DTRN_gemm_xp<<<gg, gb, 0, stream>>>(x1, l1Wih_f, l1bih_f, l1bhh_f, xp_f, 512);
    DTRN_gemm_xp<<<gg, gb, 0, stream>>>(x1, l1Wih_b, l1bih_b, l1bhh_b, xp_b, 512);
    float* outp = (float*)d_out;
    DTRN_scan<<<2, 512, 0, stream>>>(xp_f, xp_b, wf_1f, wf_1b,
                                     cfA, cbA, outp, outp + H_, 2 * H_);
}
